// RNNModel_88089779241354
// MI455X (gfx1250) — matrix operand order repaired and checked by reading, not compiled, not run on hardware
//
#include <hip/hip_runtime.h>

// Problem constants (match reference): B=128, T=1024, V=96, H=128
constexpr int BB = 128;
constexpr int TT = 1024;
constexpr int VV = 96;
constexpr int HH = 128;

typedef __attribute__((ext_vector_type(16))) _Float16 v16h;
typedef __attribute__((ext_vector_type(8)))  float    v8f;

union AV { v16h v; unsigned u[8]; };   // A/B matrix operand (16 f16 = 8 dwords/lane)
union CV { v8f  v; float    f[8]; };   // C/D accumulator (8 f32/lane)

__device__ inline unsigned pkh2(float a, float b) {
  union { _Float16 h[2]; unsigned u; } x;
  x.h[0] = (_Float16)a; x.h[1] = (_Float16)b;
  return x.u;
}

// Branch-free tanh: t = exp(-2|x|) in (0,1] (no overflow),
// tanh = sign(x)*(1-t)*rcp(1+t). Denominator in [1,2] => v_rcp_f32 is safe.
// Lowers to v_exp_f32 + v_rcp_f32 + ~3 VALU ops, no EXEC divergence, no IEEE div.
__device__ inline float fast_tanh(float x) {
  float t = __expf(-2.0f * fabsf(x));
  float r = (1.0f - t) * __builtin_amdgcn_rcpf(1.0f + t);
  return copysignf(r, x);
}

// One workgroup = one 16-row batch tile, 8 wave32s, persistent over all T steps.
__global__ __launch_bounds__(256) void rnn_char_fused(
    const int*   __restrict__ char_in,   // [B,T]
    const float* __restrict__ W_x,       // [V,H]
    const float* __restrict__ W_h,       // [H,H]
    const float* __restrict__ b_h,       // [H]
    const float* __restrict__ W_out,     // [H,V]
    const float* __restrict__ b_out,     // [V]
    const float* __restrict__ h0,        // [B,H]
    float* __restrict__ probs,           // [B,T,V]
    float* __restrict__ final_state,     // [B,H]
    float* __restrict__ amax)            // [B,T] (argmax as float)
{
  __shared__ float    sWx[VV * HH];      // 48 KB: W_x rows for gather-init of C
  __shared__ float    sbh[HH];
  __shared__ float    sbo[VV];
  __shared__ unsigned sh16[16 * 64];     // 4 KB: h tile as f16 [16][128]
  __shared__ float    slog[16 * VV];     // 6 KB: logits tile for softmax
  __shared__ int      sIdx[2][16];       // double-buffered char indices

  const int tid  = threadIdx.x;
  const int wave = tid >> 5;
  const int lane = tid & 31;
  const int lh   = lane & 15;            // lane within half-wave
  const int hi   = lane >> 4;            // half-wave select
  const int b0   = blockIdx.x * 16;      // batch tile base
  const int n0   = wave * 16;            // this wave's N-tile base

  // ---------------- preload LDS ----------------
  for (int i = tid; i < VV * HH; i += 256) sWx[i] = W_x[i];
  if (tid < HH) sbh[tid] = b_h[tid];
  if (tid < VV) sbo[tid] = b_out[tid];
  {
    _Float16* sh = (_Float16*)sh16;
    for (int i = tid; i < 16 * HH; i += 256) {
      int m = i >> 7, k = i & (HH - 1);
      sh[m * HH + k] = (_Float16)h0[(size_t)(b0 + m) * HH + k];
    }
  }
  if (tid < 16) sIdx[0][tid] = char_in[(size_t)(b0 + tid) * TT + 0];

  // ------- pack W_h / W_out into registers in a contiguous-k WMMA B layout (same k order as loadA) -------
  // B (K x 16, f16): lanes 0-15 hold K=0..15 of a 32-chunk (pairs in dwords 0..7),
  // lanes 16-31 hold K=16..31; N = n0 + (lane&15).
  AV bh4[4];
  for (int kc = 0; kc < 4; ++kc)
    for (int j = 0; j < 8; ++j) {
      int k = kc * 32 + hi * 16 + 2 * j;
      bh4[kc].u[j] = pkh2(W_h[(size_t)k * HH + n0 + lh],
                          W_h[(size_t)(k + 1) * HH + n0 + lh]);
    }
  AV bo4[4];
  if (wave < 6) {                        // V = 96 = 6 tiles of 16
    for (int kc = 0; kc < 4; ++kc)
      for (int j = 0; j < 8; ++j) {
        int k = kc * 32 + hi * 16 + 2 * j;
        bo4[kc].u[j] = pkh2(W_out[(size_t)k * VV + n0 + lh],
                            W_out[(size_t)(k + 1) * VV + n0 + lh]);
      }
  }
  __syncthreads();

  // Loop-invariant biases for this wave's N-tile (keep off the per-step path)
  const float bhn = sbh[n0 + lh];
  const float bon = (wave < 6) ? sbo[n0 + lh] : 0.0f;

  // A-matrix (16 x 32 f16) load from LDS h tile, contiguous-k order (same as B packing):
  // lanes 0-15: M=lane, K pairs {0..7} (K=0..15); lanes 16-31: pairs {8..15} (K=16..31)
  auto loadA = [&](AV a[4]) {
    for (int kc = 0; kc < 4; ++kc)
      for (int j = 0; j < 8; ++j) {
        int off = kc * 16 + 8 * hi + j;
        a[kc].u[j] = sh16[lh * 64 + off];
      }
  };

  AV a4[4];
  loadA(a4);                             // A = h0 tile

#pragma unroll 1
  for (int t = 0; t < TT; ++t) {
    // ---- C init = xproj = W_x[char] + b_h (gather replaces one_hot @ W_x) ----
    CV c;
    for (int r = 0; r < 8; ++r) {
      int m = r + 8 * hi;                // C/D layout: VGPR r -> row r (+8 for hi)
      c.f[r] = sWx[sIdx[t & 1][m] * HH + n0 + lh] + bhn;
    }
    // ---- h_{t-1} @ W_h via 4 chained WMMAs (K = 128) ----
    c.v = __builtin_amdgcn_wmma_f32_16x16x32_f16(false, a4[0].v, false, bh4[0].v, (short)0, c.v, false, false);
    c.v = __builtin_amdgcn_wmma_f32_16x16x32_f16(false, a4[1].v, false, bh4[1].v, (short)0, c.v, false, false);
    c.v = __builtin_amdgcn_wmma_f32_16x16x32_f16(false, a4[2].v, false, bh4[2].v, (short)0, c.v, false, false);
    c.v = __builtin_amdgcn_wmma_f32_16x16x32_f16(false, a4[3].v, false, bh4[3].v, (short)0, c.v, false, false);
    // ---- activation (branch-free: v_exp_f32 + v_rcp_f32) ----
    for (int r = 0; r < 8; ++r) c.f[r] = fast_tanh(c.f[r]);

    __syncthreads();  // all waves done reading previous h / logits LDS

    // ---- publish h_t (f16) + fetch next step's indices ----
    {
      _Float16* sh = (_Float16*)sh16;
      for (int r = 0; r < 8; ++r)
        sh[(r + 8 * hi) * HH + n0 + lh] = (_Float16)c.f[r];
    }
    if (tid < 16 && t + 1 < TT)
      sIdx[(t + 1) & 1][tid] = char_in[(size_t)(b0 + tid) * TT + (t + 1)];
    if (t == TT - 1) {                   // final state straight from f32 regs
      for (int r = 0; r < 8; ++r)
        final_state[(size_t)(b0 + r + 8 * hi) * HH + n0 + lh] = c.f[r];
    }
    __syncthreads();

    // ---- reload A = h_t: feeds logits now AND next step's recurrence ----
    loadA(a4);

    // ---- logits tile = h_t @ W_out + b_out (waves 0..5 cover V=96) ----
    if (wave < 6) {
      CV lg;
      for (int r = 0; r < 8; ++r) lg.f[r] = bon;
      lg.v = __builtin_amdgcn_wmma_f32_16x16x32_f16(false, a4[0].v, false, bo4[0].v, (short)0, lg.v, false, false);
      lg.v = __builtin_amdgcn_wmma_f32_16x16x32_f16(false, a4[1].v, false, bo4[1].v, (short)0, lg.v, false, false);
      lg.v = __builtin_amdgcn_wmma_f32_16x16x32_f16(false, a4[2].v, false, bo4[2].v, (short)0, lg.v, false, false);
      lg.v = __builtin_amdgcn_wmma_f32_16x16x32_f16(false, a4[3].v, false, bo4[3].v, (short)0, lg.v, false, false);
      for (int r = 0; r < 8; ++r)
        slog[(r + 8 * hi) * VV + n0 + lh] = lg.f[r];
    }
    __syncthreads();

    // ---- softmax + argmax: 16 threads per row, 6 cols each, shfl reductions ----
    {
      int row = tid >> 4, sub = tid & 15;
      float vals[6];
      float vmax = -3.4e38f; int vidx = 0;
      for (int i = 0; i < 6; ++i) {
        float x = slog[row * VV + sub + 16 * i];
        vals[i] = x;
        if (x > vmax) { vmax = x; vidx = sub + 16 * i; }
      }
      for (int off = 8; off; off >>= 1) {       // max with first-occurrence ties
        float om = __shfl_xor(vmax, off, 16);
        int   oi = __shfl_xor(vidx, off, 16);
        if (om > vmax || (om == vmax && oi < vidx)) { vmax = om; vidx = oi; }
      }
      float s = 0.f;
      for (int i = 0; i < 6; ++i) { vals[i] = __expf(vals[i] - vmax); s += vals[i]; }
      for (int off = 8; off; off >>= 1) s += __shfl_xor(s, off, 16);
      float inv = __builtin_amdgcn_rcpf(s);     // s >= 1, v_rcp_f32 is safe
      size_t pb = ((size_t)(b0 + row) * TT + t) * VV;
      for (int i = 0; i < 6; ++i) probs[pb + sub + 16 * i] = vals[i] * inv;
      if (sub == 0) amax[(size_t)(b0 + row) * TT + t] = (float)vidx;
    }
    // next iter's first barrier orders these LDS reads vs. future writes
  }
}

extern "C" void kernel_launch(void* const* d_in, const int* in_sizes, int n_in,
                              void* d_out, int out_size, void* d_ws, size_t ws_size,
                              hipStream_t stream) {
  const int*   char_in = (const int*)  d_in[0];
  const float* W_x     = (const float*)d_in[1];
  const float* W_h     = (const float*)d_in[2];
  const float* b_h     = (const float*)d_in[3];
  const float* W_out   = (const float*)d_in[4];
  const float* b_out   = (const float*)d_in[5];
  const float* h0      = (const float*)d_in[6];

  float* out         = (float*)d_out;
  float* probs       = out;                                   // B*T*V
  float* final_state = out + (size_t)BB * TT * VV;            // B*H
  float* amax        = final_state + (size_t)BB * HH;         // B*T

  rnn_char_fused<<<BB / 16, 256, 0, stream>>>(
      char_in, W_x, W_h, b_h, W_out, b_out, h0, probs, final_state, amax);
}
